// SelfAttention_69879117906503
// MI455X (gfx1250) — matrix operand order repaired and checked by reading, not compiled, not run on hardware
//
#include <hip/hip_runtime.h>

// Causal self-attention forward for B=2,H=16,S=2048,D=64 (fp32 in/out).
// d_out = O [B,H,S,D] ++ attention_weights [B,H,S,S], both fp32.
// Per-block = one 16-row query tile of one (b,h). WMMA f16 GEMMs, f32 softmax.
// Pass 1: QK^T (WMMA) -> raw f16 scores stashed in LDS + per-lane row-max.
// Pass 2: LDS-only exp + row-sum (no recompute, no global traffic).
// Pass 3: P@V (WMMA from the LDS stash), 1/rowsum folded into O.
// Weights (537 MB, the dominant traffic) streamed with non-temporal stores.

#define S_LEN   2048
#define D_DIM   64
#define NW      4        // waves per block (128 threads)
#define PSTRIDE 2056     // 2048 + 8 halves pad -> row stride = 1028 dwords = 4 mod 64 banks

typedef __attribute__((ext_vector_type(16))) _Float16 v16h;
typedef __attribute__((ext_vector_type(8)))  float    v8f;

// A-fragment (16x32 f16): lane = h*16 + m ; element e -> K = 16h + e
// Q is pre-scaled by `s` before f32->f16 conversion (folds the 1/sqrt(D) scale).
__device__ __forceinline__ v16h pack_A_scaled(const float* __restrict__ row, int hh, float s) {
  const float4* p0 = reinterpret_cast<const float4*>(row + 16 * hh);
  const float4* p1 = reinterpret_cast<const float4*>(row + 16 * hh + 8);
  float4 f0 = p0[0], f1 = p0[1], f2 = p1[0], f3 = p1[1];
  v16h a;
  a[0] = (_Float16)(f0.x * s); a[1] = (_Float16)(f0.y * s);
  a[2] = (_Float16)(f0.z * s); a[3] = (_Float16)(f0.w * s);
  a[4] = (_Float16)(f1.x * s); a[5] = (_Float16)(f1.y * s);
  a[6] = (_Float16)(f1.z * s); a[7] = (_Float16)(f1.w * s);
  a[8]  = (_Float16)(f2.x * s); a[9]  = (_Float16)(f2.y * s);
  a[10] = (_Float16)(f2.z * s); a[11] = (_Float16)(f2.w * s);
  a[12] = (_Float16)(f3.x * s); a[13] = (_Float16)(f3.y * s);
  a[14] = (_Float16)(f3.z * s); a[15] = (_Float16)(f3.w * s);
  return a;
}

// B-fragment (32x16 f16): lane = h*16 + n ; element e -> K = 16h + e
// "row" points at 16 consecutive f32 holding K=16h..16h+15 for this lane's column.
__device__ __forceinline__ v16h pack_B_contig(const float* __restrict__ row, int hh) {
  const float4* p = reinterpret_cast<const float4*>(row + 16 * hh);
  float4 f0 = p[0], f1 = p[1], f2 = p[2], f3 = p[3];
  v16h b;
  b[0] = (_Float16)f0.x; b[1] = (_Float16)f0.y; b[2] = (_Float16)f0.z; b[3] = (_Float16)f0.w;
  b[4] = (_Float16)f1.x; b[5] = (_Float16)f1.y; b[6] = (_Float16)f1.z; b[7] = (_Float16)f1.w;
  b[8]  = (_Float16)f2.x; b[9]  = (_Float16)f2.y; b[10] = (_Float16)f2.z; b[11] = (_Float16)f2.w;
  b[12] = (_Float16)f3.x; b[13] = (_Float16)f3.y; b[14] = (_Float16)f3.z; b[15] = (_Float16)f3.w;
  return b;
}

__global__ __launch_bounds__(128) void attn_fwd_kernel(
    const float* __restrict__ Q, const float* __restrict__ Km,
    const float* __restrict__ V, const float* __restrict__ scale_p,
    float* __restrict__ O, float* __restrict__ W)
{
  __shared__ _Float16 Pbuf[16][PSTRIDE];   // pass1: raw f16 scores; pass2+: exp(score-rowmax)
  __shared__ float redmax[NW][16];
  __shared__ float redsum[NW][16];
  __shared__ float rowinv[16];             // 1 / rowsum

  const int tid  = threadIdx.x;
  const int wid  = tid >> 5;
  const int lane = tid & 31;
  const int hh   = lane >> 4;    // half-wave index
  const int l16  = lane & 15;    // A-row m / B-col n / C-col n
  const int qt   = blockIdx.x;   // query tile (16 rows)
  const int bh   = blockIdx.y;   // fused batch*head

  const float scale = scale_p[0];

  // ---- zero the P stash (future/masked tiles stay 0 -> weights exactly 0) ----
  {
    unsigned int* pz = (unsigned int*)&Pbuf[0][0];
    for (int i = tid; i < 16 * PSTRIDE / 2; i += 128) pz[i] = 0u;
  }

  // ---- Q tile A-fragments, pre-scaled ----
  const float* qrow = Q + ((size_t)bh * S_LEN + (size_t)qt * 16 + l16) * D_DIM;
  const v16h aq0 = pack_A_scaled(qrow, hh, scale);        // K-dim 0..31
  const v16h aq1 = pack_A_scaled(qrow + 32, hh, scale);   // K-dim 32..63

  __syncthreads();   // zero-init visible before any wave writes score tiles

  // ========== Pass 1: QK^T (WMMA), stash raw f16 scores, per-lane row-max ======
  float rm[8];
#pragma unroll
  for (int j = 0; j < 8; ++j) rm[j] = -1e30f;

  const float* kbase = Km + (size_t)bh * S_LEN * D_DIM + (size_t)l16 * D_DIM;
  for (int kt = wid; kt <= qt; kt += NW) {
    const float* krow = kbase + (size_t)kt * 16 * D_DIM;
    if (kt + NW <= qt) {  // prefetch next owned K tile (global_prefetch_b8)
      const float* knext = krow + (size_t)NW * 16 * D_DIM;
      __builtin_prefetch(knext, 0, 1);
      __builtin_prefetch(knext + 32, 0, 1);
    }
    v16h bk0 = pack_B_contig(krow, hh);
    v16h bk1 = pack_B_contig(krow + 32, hh);
    v8f c = {};
    c = __builtin_amdgcn_wmma_f32_16x16x32_f16(false, aq0, false, bk0, (short)0, c, false, false);
    c = __builtin_amdgcn_wmma_f32_16x16x32_f16(false, aq1, false, bk1, (short)0, c, false, false);
    const bool diag = (kt == qt);
#pragma unroll
    for (int j = 0; j < 8; ++j) {
      const int M = j + 8 * hh;
      const float sv = (!diag || (l16 <= M)) ? c[j] : -1e30f;
      rm[j] = fmaxf(rm[j], sv);
      Pbuf[M][kt * 16 + l16] = (_Float16)c[j];   // raw score; mask applied in pass 2
    }
  }
  // one cross-lane max reduce per row (width-16 keeps the two halves separate)
#pragma unroll
  for (int j = 0; j < 8; ++j) {
    float v = rm[j];
#pragma unroll
    for (int m = 8; m >= 1; m >>= 1) v = fmaxf(v, __shfl_xor(v, m, 16));
    rm[j] = v;
  }
  if (l16 == 0) {
#pragma unroll
    for (int j = 0; j < 8; ++j) redmax[wid][j + 8 * hh] = rm[j];
  }
  __syncthreads();

  float rowM[8];
#pragma unroll
  for (int j = 0; j < 8; ++j) {
    float v = redmax[0][j + 8 * hh];
#pragma unroll
    for (int w = 1; w < NW; ++w) v = fmaxf(v, redmax[w][j + 8 * hh]);
    rowM[j] = v;
  }

  // ========== Pass 2: LDS-only exp + row sum (in-place score -> p) =============
  float rs[8];
#pragma unroll
  for (int j = 0; j < 8; ++j) rs[j] = 0.0f;

  for (int kt = wid; kt <= qt; kt += NW) {  // same tiles this wave wrote in pass 1
    const bool diag = (kt == qt);
#pragma unroll
    for (int j = 0; j < 8; ++j) {
      const int M = j + 8 * hh;
      const bool valid = (!diag || (l16 <= M));
      const float s = (float)Pbuf[M][kt * 16 + l16];
      const float p = valid ? __expf(s - rowM[j]) : 0.0f;
      rs[j] += p;                      // per-lane partial; reduce once after loop
      Pbuf[M][kt * 16 + l16] = (_Float16)p;
    }
  }
#pragma unroll
  for (int j = 0; j < 8; ++j) {
    float v = rs[j];
#pragma unroll
    for (int m = 8; m >= 1; m >>= 1) v += __shfl_xor(v, m, 16);
    rs[j] = v;
  }
  if (l16 == 0) {
#pragma unroll
    for (int j = 0; j < 8; ++j) redsum[wid][j + 8 * hh] = rs[j];
  }
  __syncthreads();
  if (tid < 16) {
    float s = redsum[0][tid] + redsum[1][tid] + redsum[2][tid] + redsum[3][tid];
    rowinv[tid] = 1.0f / s;   // diagonal guarantees s >= 1
  }
  __syncthreads();

  // ========== Pass 3: O = (P @ V) * (1/rowsum) ================================
  // wave `wid` owns output columns d = wid*16 + l16
  {
    v8f co = {};
    const int d = wid * 16 + l16;
    const float* vbase = V + (size_t)bh * S_LEN * D_DIM + d;
    const int nchunks = (qt + 2) >> 1;   // 32-wide K chunks covering (qt+1)*16 keys
    for (int c2 = 0; c2 < nchunks; ++c2) {
      v16h a;
      const _Float16* pr = &Pbuf[l16][c2 * 32 + 16 * hh];
#pragma unroll
      for (int e = 0; e < 8; ++e) { a[e] = pr[e]; a[8 + e] = pr[8 + e]; }
      v16h b;
      const float* vr = vbase + (size_t)(c2 * 32 + 16 * hh) * D_DIM;
#pragma unroll
      for (int e = 0; e < 16; ++e) b[e] = (_Float16)vr[(size_t)e * D_DIM];
      co = __builtin_amdgcn_wmma_f32_16x16x32_f16(false, a, false, b, (short)0, co, false, false);
    }
#pragma unroll
    for (int j = 0; j < 8; ++j) {
      const int M = j + 8 * hh;
      O[((size_t)bh * S_LEN + (size_t)qt * 16 + M) * D_DIM + d] = co[j] * rowinv[M];
    }
  }

  // ========== Stream normalized weights (dominant traffic, NT stores) ==========
  {
    float* wbase = W + ((size_t)bh * S_LEN + (size_t)qt * 16) * S_LEN;
    for (int idx = tid; idx < 16 * S_LEN; idx += 128) {
      const int r  = idx >> 11;          // / 2048
      const int cc = idx & (S_LEN - 1);  // % 2048
      const float w = (float)Pbuf[r][cc] * rowinv[r];
      __builtin_nontemporal_store(w, &wbase[(size_t)r * S_LEN + cc]);
    }
  }
}

extern "C" void kernel_launch(void* const* d_in, const int* in_sizes, int n_in,
                              void* d_out, int out_size, void* d_ws, size_t ws_size,
                              hipStream_t stream) {
  (void)in_sizes; (void)n_in; (void)out_size; (void)d_ws; (void)ws_size;
  const float* Q     = (const float*)d_in[0];
  const float* K     = (const float*)d_in[1];
  const float* V     = (const float*)d_in[2];
  const float* scale = (const float*)d_in[3];
  // d_in[4] is the causal mask; causality is implemented analytically.

  float* O = (float*)d_out;
  float* W = O + (size_t)2 * 16 * S_LEN * D_DIM;  // weights follow O in d_out

  dim3 grid(S_LEN / 16, 2 * 16);  // 128 query tiles x (B*H = 32)
  attn_fwd_kernel<<<grid, 128, 0, stream>>>(Q, K, V, scale, O, W);
}
